// LlamaAttention_61899068670751
// MI455X (gfx1250) — matrix operand order repaired and checked by reading, not compiled, not run on hardware
//
#include <hip/hip_runtime.h>
#include <hip/hip_bf16.h>

#define S_LEN 2048
#define NH    32
#define NKV   8
#define HD    128
#define HIDN  4096
#define NQKV  6144   // H*D + 2*KV*D
#define SINKW 128
#define WINW  1024
#define POOLN 100

// CDNA5 async global->LDS copies (ASYNCcnt path). Set to 0 to fall back to
// the load/ds_store path.
#define USE_ASYNC_LDS 1

typedef __attribute__((ext_vector_type(16))) __bf16 v16bf;
typedef __attribute__((ext_vector_type(8)))  float  v8f;

typedef union { v16bf v; uint4 u[2]; } frag16;

__device__ __forceinline__ unsigned short f2bf(float f) {
  unsigned int u = __float_as_uint(f);
  u += 0x7FFFu + ((u >> 16) & 1u);          // round-to-nearest-even
  return (unsigned short)(u >> 16);
}

__device__ __forceinline__ v8f wmma_bf16(const frag16& a, const frag16& b, v8f c) {
  return __builtin_amdgcn_wmma_f32_16x16x32_bf16(false, a.v, false, b.v, (short)0, c, false, false);
}

#if USE_ASYNC_LDS
// One instruction moves 16B per lane (512B per wave) memory -> LDS, no VGPR staging.
// %0 = per-lane LDS byte offset (low 32 bits of flat shared address == LDS offset),
// %1 = per-lane 64-bit global address.
__device__ __forceinline__ void async_b128(unsigned lds_byte_off, const void* gaddr) {
  asm volatile("global_load_async_to_lds_b128 %0, %1, off"
               :: "v"(lds_byte_off), "v"(gaddr) : "memory");
}
__device__ __forceinline__ void async_wait0() {
  asm volatile("s_wait_asynccnt 0x0" ::: "memory");
}
#endif

// ---------------------------------------------------------------- converters
__global__ void k_cvt_x(const float* __restrict__ x, unsigned short* __restrict__ xb, int n) {
  int i = blockIdx.x * blockDim.x + threadIdx.x;
  if (i < n) xb[i] = f2bf(x[i]);
}

__global__ void k_pack_wqkv(const float* __restrict__ Wq, const float* __restrict__ Wk,
                            const float* __restrict__ Wv, unsigned short* __restrict__ Wt) {
  size_t idx = (size_t)blockIdx.x * blockDim.x + threadIdx.x;
  if (idx >= (size_t)NQKV * HIDN) return;
  int n = (int)(idx / HIDN);
  int k = (int)(idx % HIDN);
  float v;
  if (n < HIDN)             v = Wq[(size_t)k * HIDN + n];
  else if (n < HIDN + 1024) v = Wk[(size_t)k * 1024 + (n - HIDN)];
  else                      v = Wv[(size_t)k * 1024 + (n - HIDN - 1024)];
  Wt[idx] = f2bf(v);
}

__global__ void k_pack_wo(const float* __restrict__ Wo, unsigned short* __restrict__ Wt) {
  size_t idx = (size_t)blockIdx.x * blockDim.x + threadIdx.x;
  if (idx >= (size_t)HIDN * HIDN) return;
  int n = (int)(idx / HIDN);
  int k = (int)(idx % HIDN);
  Wt[idx] = f2bf(Wo[(size_t)k * HIDN + n]);
}

// ---------------------------------------------------------------- bf16 GEMM
// C[M][N] = A[M][K] * Bt[N][K]^T ; M,N multiples of 128, K multiple of 32.
// 128x128 block, 8 waves (2M x 4N), wave tile 64x32, K staged 32-wide through
// double-buffered LDS filled by async global->LDS DMA (one barrier per step).
__global__ __launch_bounds__(256)
void k_gemm_bf16(const unsigned short* __restrict__ A,
                 const unsigned short* __restrict__ Bt,
                 float* __restrict__ C, int M, int N, int K) {
  // two buffers each: 128 rows x 32 halves padded to 40 (bank-conflict-free)
  __shared__ __attribute__((aligned(16))) unsigned short Ash[2 * 128 * 40];
  __shared__ __attribute__((aligned(16))) unsigned short Bsh[2 * 128 * 40];
  (void)M;
  const int tid  = threadIdx.x;
  const int lane = tid & 31, wid = tid >> 5;
  const int wm = wid & 1, wn = wid >> 1;          // 2 (M) x 4 (N) waves
  const int nl = lane & 15, half = lane >> 4;
  const int bx = blockIdx.x, by = blockIdx.y;

  v8f acc[4][2];
  v8f zero = {0.f,0.f,0.f,0.f,0.f,0.f,0.f,0.f};
#pragma unroll
  for (int i = 0; i < 4; ++i) { acc[i][0] = zero; acc[i][1] = zero; }

  const unsigned short* Abase = A  + (size_t)(by * 128) * K;
  const unsigned short* Bbase = Bt + (size_t)(bx * 128) * K;

#if USE_ASYNC_LDS
  const unsigned shA0 = (unsigned)(uintptr_t)&Ash[0];
  const unsigned shB0 = (unsigned)(uintptr_t)&Bsh[0];
  // tile = 128 rows x 64B = 512 chunks of 16B; 2 chunks per thread per matrix
  auto issue_tile = [&](int kk, int buf) {
#pragma unroll
    for (int i = 0; i < 2; ++i) {
      int chunk = i * 256 + tid;
      int row = chunk >> 2, c4 = chunk & 3;
      unsigned loff = (unsigned)(row * 80 + c4 * 32) + (unsigned)(buf * 10240);
      const unsigned short* ga = Abase + (size_t)row * K + kk + c4 * 16;
      const unsigned short* gb = Bbase + (size_t)row * K + kk + c4 * 16;
      async_b128(shA0 + loff, ga);
      async_b128(shB0 + loff, gb);
    }
  };
  issue_tile(0, 0);
#else
  const int lrow = tid >> 1, lseg = tid & 1;
  const unsigned short* aRow = Abase + (size_t)lrow * K + lseg * 16;
  const unsigned short* bRow = Bbase + (size_t)lrow * K + lseg * 16;
  unsigned short* aDst = &Ash[lrow * 40 + lseg * 16];
  unsigned short* bDst = &Bsh[lrow * 40 + lseg * 16];
#endif

  for (int k0 = 0; k0 < K; k0 += 32) {
#if USE_ASYNC_LDS
    const int cur = (k0 >> 5) & 1;
    async_wait0();                 // my wave's fills for tile `cur` are done
    __syncthreads();               // everyone's fills done + prev compute done
    if (k0 + 32 < K) issue_tile(k0 + 32, cur ^ 1);   // DMA overlaps WMMAs
    const unsigned short* As = &Ash[cur * 5120];
    const unsigned short* Bs = &Bsh[cur * 5120];
#else
    uint4 a0 = *(const uint4*)(aRow + k0);
    uint4 a1 = *(const uint4*)(aRow + k0 + 8);
    uint4 b0 = *(const uint4*)(bRow + k0);
    uint4 b1 = *(const uint4*)(bRow + k0 + 8);
    if (k0 + 32 < K) {
      __builtin_prefetch(aRow + k0 + 32, 0, 1);
      __builtin_prefetch(bRow + k0 + 32, 0, 1);
    }
    __syncthreads();
    *(uint4*)(aDst) = a0; *(uint4*)(aDst + 8) = a1;
    *(uint4*)(bDst) = b0; *(uint4*)(bDst + 8) = b1;
    __syncthreads();
    const unsigned short* As = Ash;
    const unsigned short* Bs = Bsh;
#endif

    frag16 bf[2];
#pragma unroll
    for (int ni = 0; ni < 2; ++ni) {
      const unsigned short* p = &Bs[(wn * 32 + ni * 16 + nl) * 40];
      bf[ni].u[0] = *(const uint4*)(p + half * 8);
      bf[ni].u[1] = *(const uint4*)(p + 16 + half * 8);
    }
#pragma unroll
    for (int mi = 0; mi < 4; ++mi) {
      frag16 af;
      const unsigned short* p = &As[(wm * 64 + mi * 16 + nl) * 40];
      af.u[0] = *(const uint4*)(p + half * 8);
      af.u[1] = *(const uint4*)(p + 16 + half * 8);
      acc[mi][0] = wmma_bf16(af, bf[0], acc[mi][0]);
      acc[mi][1] = wmma_bf16(af, bf[1], acc[mi][1]);
    }
  }

#pragma unroll
  for (int mi = 0; mi < 4; ++mi)
#pragma unroll
    for (int ni = 0; ni < 2; ++ni)
#pragma unroll
      for (int r = 0; r < 8; ++r) {
        int row = by * 128 + wm * 64 + mi * 16 + r + 8 * half;
        int col = bx * 128 + wn * 32 + ni * 16 + nl;
        C[(size_t)row * N + col] = acc[mi][ni][r];
      }
}

// ---------------------------------------------------------------- RoPE + packing
// qkv f32 [S][6144] -> qb bf16 [S][H][D] (pre-scaled by 1/sqrt(D)),
//                      kb bf16 [S][KV][D], vt bf16 [KV][D][S] (transposed V)
__global__ void k_rope_pack(const float* __restrict__ qkv,
                            unsigned short* __restrict__ qb,
                            unsigned short* __restrict__ kb,
                            unsigned short* __restrict__ vt) {
  size_t idx = (size_t)blockIdx.x * blockDim.x + threadIdx.x;
  if (idx >= (size_t)S_LEN * NQKV) return;
  int s = (int)(idx / NQKV);
  int r = (int)(idx % NQKV);
  const float* row = qkv + (size_t)s * NQKV;
  if (r < HIDN + 1024) {                       // q or k: RoPE
    int loc = (r < HIDN) ? r : (r - HIDN);
    int d = loc & 127;
    int i = d & 63;
    float th = (float)s * __expf(-(float)i * 0.14391157f); // ln(1e4)/64
    float sn, c; __sincosf(th, &sn, &c);
    float val;
    if (d < 64) val = row[r] * c - row[r + 64] * sn;
    else        val = row[r] * c + row[r - 64] * sn;
    if (r < HIDN) qb[(size_t)s * HIDN + loc] = f2bf(val * 0.0883883476f); // 1/sqrt(128)
    else          kb[(size_t)s * 1024 + loc] = f2bf(val);
  } else {                                     // v: transpose to [KV][D][S]
    int loc = r - (HIDN + 1024);
    int kvh = loc >> 7, d = loc & 127;
    vt[((size_t)kvh * HD + d) * S_LEN + s] = f2bf(row[r]);
  }
}

// ---------------------------------------------------------------- router MLP
__global__ __launch_bounds__(256)
void k_router(const float* __restrict__ qkv,
              const float* fe1w, const float* fe1b,
              const float* fe2w, const float* fe2b,
              const float* r1w, const float* r1b,
              const float* r2w, const float* r2b,
              const float* r3w, const float* r3b,
              const float* noise, float* __restrict__ mix_out) {
  __shared__ float part[256];
  __shared__ float feat[128];
  __shared__ float h1[1024];
  __shared__ float h2[256];
  __shared__ float h3[512];
  __shared__ float h4[128];
  int t = threadIdx.x;
  int d = t & 127, seg = t >> 7;
  int i = d & 63;
  float acc = 0.f;
  int sbase = seg ? (S_LEN - POOLN) : 0;
  for (int sp = 0; sp < POOLN; ++sp) {
    int s = sbase + sp;
    float th = (float)s * __expf(-(float)i * 0.14391157f);
    float sn, c; __sincosf(th, &sn, &c);
    const float* row = qkv + (size_t)s * NQKV;
    for (int h = 0; h < NH; ++h) {
      const float* qh = row + h * HD;
      float val = (d < 64) ? (qh[d] * c - qh[d + 64] * sn)
                           : (qh[d] * c + qh[d - 64] * sn);
      acc += val;
    }
  }
  part[t] = acc;
  __syncthreads();
  if (t < 128) feat[t] = (part[t] + part[t + 128]) * (1.f / (2.f * POOLN * NH));
  __syncthreads();
  for (int n = t; n < 1024; n += 256) {
    float a = fe1b[n];
    for (int k = 0; k < 128; ++k) a += feat[k] * fe1w[(size_t)k * 1024 + n];
    h1[n] = a / (1.f + __expf(-a));            // silu
  }
  __syncthreads();
  if (t < 256) {
    float a = fe2b[t];
    for (int k = 0; k < 1024; ++k) a += h1[k] * fe2w[(size_t)k * 256 + t];
    h2[t] = a;
  }
  __syncthreads();
  for (int n = t; n < 512; n += 256) {
    float a = r1b[n];
    for (int k = 0; k < 256; ++k) a += h2[k] * r1w[(size_t)k * 512 + n];
    h3[n] = a / (1.f + __expf(-a));
  }
  __syncthreads();
  if (t < 128) {
    float a = r2b[t];
    for (int k = 0; k < 512; ++k) a += h3[k] * r2w[(size_t)k * 128 + t];
    h4[t] = a / (1.f + __expf(-a));
  }
  __syncthreads();
  if (t == 0) {
    float logit = r3b[0];
    for (int k = 0; k < 128; ++k) logit += h4[k] * r3w[k];
    const float eps = 1e-8f;
    float g = -__logf(-__logf(noise[0] + eps) + eps);
    float z = 1.f / (1.f + __expf(-(logit + g)));  // TAU = 1
    mix_out[0] = (z > 0.5f) ? 1.f : 0.f;           // forward value of STE
  }
}

// ---------------------------------------------------------------- dual-mask flash attention
__global__ __launch_bounds__(256)
void k_attention(const unsigned short* __restrict__ qb,   // [S][H][D], pre-scaled
                 const unsigned short* __restrict__ kb,   // [S][KV][D]
                 const unsigned short* __restrict__ vt,   // [KV][D][S]
                 const float* __restrict__ mix_p,
                 unsigned short* __restrict__ ob) {       // [S][H][D]
  __shared__ __attribute__((aligned(16))) unsigned short Plds[8 * 16 * 40];
  const int tid = threadIdx.x;
  const int lane = tid & 31, wid = tid >> 5;
  const int nl = lane & 15, half = lane >> 4;
  const int h = blockIdx.y;
  const int kvh = h >> 2;                        // H/KV = 4
  const int qs0 = blockIdx.x * 128 + wid * 16;   // wave-private 16 query rows
  unsigned short* Pw = &Plds[wid * 640];

  // Q fragments (16 rows x 128 d, 4 k-chunks of 32)
  frag16 qf[4];
  const unsigned short* qp = qb + ((size_t)(qs0 + nl) * NH + h) * HD;
#pragma unroll
  for (int c = 0; c < 4; ++c) {
    qf[c].u[0] = *(const uint4*)(qp + c * 32 + half * 8);
    qf[c].u[1] = *(const uint4*)(qp + c * 32 + 16 + half * 8);
  }

  float mF[8], lF[8], mS[8], lS[8];
  v8f oF[8], oS[8];
  v8f zero = {0.f,0.f,0.f,0.f,0.f,0.f,0.f,0.f};
#pragma unroll
  for (int r = 0; r < 8; ++r) { mF[r] = -1e30f; lF[r] = 0.f; mS[r] = -1e30f; lS[r] = 0.f; }
#pragma unroll
  for (int c = 0; c < 8; ++c) { oF[c] = zero; oS[c] = zero; }

  for (int j0 = 0; j0 <= qs0 + 15; j0 += 32) {
    // ---- scores: S = q @ K^T  (two 16-key halves)
    v8f s0 = zero, s1 = zero;
#pragma unroll
    for (int c = 0; c < 4; ++c) {
      frag16 kf0, kf1;
      const unsigned short* kp0 = kb + ((size_t)(j0 + nl) * NKV + kvh) * HD + c * 32 + half * 8;
      kf0.u[0] = *(const uint4*)kp0; kf0.u[1] = *(const uint4*)(kp0 + 16);
      const unsigned short* kp1 = kb + ((size_t)(j0 + 16 + nl) * NKV + kvh) * HD + c * 32 + half * 8;
      kf1.u[0] = *(const uint4*)kp1; kf1.u[1] = *(const uint4*)(kp1 + 16);
      s0 = wmma_bf16(qf[c], kf0, s0);
      s1 = wmma_bf16(qf[c], kf1, s1);
    }

    const bool strm_need = (j0 < SINKW) || (j0 + 31 > qs0 - WINW);  // wave-uniform
    const int jc0 = j0 + nl, jc1 = j0 + 16 + nl;

    // ---- full-causal softmax update + P(full) to LDS
#pragma unroll
    for (int r = 0; r < 8; ++r) {
      int irow = qs0 + r + 8 * half;
      float f0 = (jc0 <= irow) ? s0[r] : -1e30f;
      float f1 = (jc1 <= irow) ? s1[r] : -1e30f;
      float rm = fmaxf(f0, f1);
      rm = fmaxf(rm, __shfl_xor(rm, 1)); rm = fmaxf(rm, __shfl_xor(rm, 2));
      rm = fmaxf(rm, __shfl_xor(rm, 4)); rm = fmaxf(rm, __shfl_xor(rm, 8));
      float mnew = fmaxf(mF[r], rm);
      float scl = __expf(mF[r] - mnew);
      float p0 = __expf(f0 - mnew), p1 = __expf(f1 - mnew);
      float ps = p0 + p1;
      ps += __shfl_xor(ps, 1); ps += __shfl_xor(ps, 2);
      ps += __shfl_xor(ps, 4); ps += __shfl_xor(ps, 8);
      lF[r] = lF[r] * scl + ps;
      mF[r] = mnew;
#pragma unroll
      for (int c = 0; c < 8; ++c) oF[c][r] = oF[c][r] * scl;
      Pw[(r + 8 * half) * 40 + nl]      = f2bf(p0);
      Pw[(r + 8 * half) * 40 + 16 + nl] = f2bf(p1);
    }
    frag16 pfF;
    {
      const unsigned short* pp = &Pw[nl * 40];
      pfF.u[0] = *(const uint4*)(pp + half * 8);
      pfF.u[1] = *(const uint4*)(pp + 16 + half * 8);
    }

    // ---- streaming-mask softmax update + P(strm) to LDS
    frag16 pfS = pfF;
    if (strm_need) {
#pragma unroll
      for (int r = 0; r < 8; ++r) {
        int irow = qs0 + r + 8 * half;
        bool a0 = (jc0 <= irow) && ((jc0 < SINKW) || (jc0 > irow - WINW));
        bool a1 = (jc1 <= irow) && ((jc1 < SINKW) || (jc1 > irow - WINW));
        float f0 = a0 ? s0[r] : -1e30f;
        float f1 = a1 ? s1[r] : -1e30f;
        float rm = fmaxf(f0, f1);
        rm = fmaxf(rm, __shfl_xor(rm, 1)); rm = fmaxf(rm, __shfl_xor(rm, 2));
        rm = fmaxf(rm, __shfl_xor(rm, 4)); rm = fmaxf(rm, __shfl_xor(rm, 8));
        float mnew = fmaxf(mS[r], rm);
        float scl = __expf(mS[r] - mnew);
        float p0 = __expf(f0 - mnew), p1 = __expf(f1 - mnew);
        float ps = p0 + p1;
        ps += __shfl_xor(ps, 1); ps += __shfl_xor(ps, 2);
        ps += __shfl_xor(ps, 4); ps += __shfl_xor(ps, 8);
        lS[r] = lS[r] * scl + ps;
        mS[r] = mnew;
#pragma unroll
        for (int c = 0; c < 8; ++c) oS[c][r] = oS[c][r] * scl;
        Pw[(r + 8 * half) * 40 + nl]      = f2bf(p0);
        Pw[(r + 8 * half) * 40 + 16 + nl] = f2bf(p1);
      }
      const unsigned short* pp = &Pw[nl * 40];
      pfS.u[0] = *(const uint4*)(pp + half * 8);
      pfS.u[1] = *(const uint4*)(pp + 16 + half * 8);
    }

    // ---- O += P @ V  (V loaded once per d-chunk, shared by both variants)
#pragma unroll
    for (int c = 0; c < 8; ++c) {
      frag16 vf;
      const unsigned short* vp = vt + ((size_t)kvh * HD + c * 16 + nl) * S_LEN + j0 + half * 8;
      vf.u[0] = *(const uint4*)vp; vf.u[1] = *(const uint4*)(vp + 16);
      oF[c] = wmma_bf16(pfF, vf, oF[c]);
      if (strm_need) oS[c] = wmma_bf16(pfS, vf, oS[c]);
    }
  }

  // ---- normalize, mix, store bf16
  const float mix = mix_p[0];
  float wF[8], wS[8];
#pragma unroll
  for (int r = 0; r < 8; ++r) { wF[r] = (1.f - mix) / lF[r]; wS[r] = mix / lS[r]; }
#pragma unroll
  for (int c = 0; c < 8; ++c)
#pragma unroll
    for (int r = 0; r < 8; ++r) {
      int srow = qs0 + r + 8 * half;
      int d = c * 16 + nl;
      float val = oF[c][r] * wF[r] + oS[c][r] * wS[r];
      ob[((size_t)srow * NH + h) * HD + d] = f2bf(val);
    }
}

// ---------------------------------------------------------------- launch
extern "C" void kernel_launch(void* const* d_in, const int* in_sizes, int n_in,
                              void* d_out, int out_size, void* d_ws, size_t ws_size,
                              hipStream_t stream) {
  (void)in_sizes; (void)n_in; (void)out_size; (void)ws_size;
  const float* hs   = (const float*)d_in[0];
  const float* Wq   = (const float*)d_in[1];
  const float* Wk   = (const float*)d_in[2];
  const float* Wv   = (const float*)d_in[3];
  const float* Wo   = (const float*)d_in[4];
  const float* fe1w = (const float*)d_in[5];
  const float* fe1b = (const float*)d_in[6];
  const float* fe2w = (const float*)d_in[7];
  const float* fe2b = (const float*)d_in[8];
  const float* r1w  = (const float*)d_in[9];
  const float* r1b  = (const float*)d_in[10];
  const float* r2w  = (const float*)d_in[11];
  const float* r2b  = (const float*)d_in[12];
  const float* r3w  = (const float*)d_in[13];
  const float* r3b  = (const float*)d_in[14];
  const float* nz   = (const float*)d_in[15];

  char* ws = (char*)d_ws;
  size_t off = 0;
  auto alloc = [&](size_t bytes) -> void* {
    void* p = ws + off;
    off = (off + bytes + 255) & ~(size_t)255;
    return p;
  };
  unsigned short* Xb  = (unsigned short*)alloc((size_t)S_LEN * HIDN * 2);
  unsigned short* Wtq = (unsigned short*)alloc((size_t)NQKV * HIDN * 2);
  unsigned short* Wto = (unsigned short*)alloc((size_t)HIDN * HIDN * 2);
  float*          QKV = (float*)alloc((size_t)S_LEN * NQKV * 4);
  unsigned short* Qb  = (unsigned short*)alloc((size_t)S_LEN * HIDN * 2);
  unsigned short* Kb  = (unsigned short*)alloc((size_t)S_LEN * NKV * HD * 2);
  unsigned short* Vt  = (unsigned short*)alloc((size_t)NKV * HD * S_LEN * 2);
  unsigned short* Ob  = (unsigned short*)alloc((size_t)S_LEN * HIDN * 2);
  float*          Mix = (float*)alloc(256);

  int nx = S_LEN * HIDN;
  k_cvt_x<<<(nx + 255) / 256, 256, 0, stream>>>(hs, Xb, nx);
  size_t nwq = (size_t)NQKV * HIDN;
  k_pack_wqkv<<<(unsigned)((nwq + 255) / 256), 256, 0, stream>>>(Wq, Wk, Wv, Wtq);
  size_t nwo = (size_t)HIDN * HIDN;
  k_pack_wo<<<(unsigned)((nwo + 255) / 256), 256, 0, stream>>>(Wo, Wto);

  // fused QKV projection: [2048 x 4096] @ [4096 x 6144]
  k_gemm_bf16<<<dim3(NQKV / 128, S_LEN / 128), 256, 0, stream>>>(Xb, Wtq, QKV, S_LEN, NQKV, HIDN);

  size_t nr = (size_t)S_LEN * NQKV;
  k_rope_pack<<<(unsigned)((nr + 255) / 256), 256, 0, stream>>>(QKV, Qb, Kb, Vt);
  k_router<<<1, 256, 0, stream>>>(QKV, fe1w, fe1b, fe2w, fe2b,
                                  r1w, r1b, r2w, r2b, r3w, r3b, nz, Mix);
  k_attention<<<dim3(S_LEN / 128, NH), 256, 0, stream>>>(Qb, Kb, Vt, Mix, Ob);

  // output projection: [2048 x 4096] @ [4096 x 4096] -> fp32 d_out
  k_gemm_bf16<<<dim3(HIDN / 128, S_LEN / 128), 256, 0, stream>>>(Ob, Wto, (float*)d_out, S_LEN, HIDN, HIDN);
}
